// GATv2_41120016892387
// MI455X (gfx1250) — matrix operand order repaired and checked by reading, not compiled, not run on hardware
//
#include <hip/hip_runtime.h>
#include <cstdint>

#define NN   100000
#define NE   1600000
#define DIN  128
#define DH   64
#define DOUT 40

typedef __attribute__((ext_vector_type(16))) __bf16 v16bf;
typedef __attribute__((ext_vector_type(8)))  float  v8f;

// ---------- order-preserving float <-> uint encoding for atomic max ----------
__device__ __forceinline__ unsigned enc_ord(float f) {
  unsigned u = __float_as_uint(f);
  return (u & 0x80000000u) ? ~u : (u | 0x80000000u);
}
__device__ __forceinline__ float dec_ord(unsigned k) {
  unsigned u = (k & 0x80000000u) ? (k & 0x7FFFFFFFu) : ~k;
  return __uint_as_float(u);
}

// ---------- init per-layer state: segment max (= -inf encoded), sum, out=0 ----------
__global__ void init_layer_k(unsigned* __restrict__ menc, float* __restrict__ s,
                             float* __restrict__ out, int n, int C) {
  int i = blockIdx.x * blockDim.x + threadIdx.x;
  if (i < n) { menc[i] = 0x007FFFFFu; /* enc(-inf) */ s[i] = 0.0f; }
  if (i < n * C) out[i] = 0.0f;
}

// ---------- pack W[K,C] (f32) into WMMA-B-operand order, bf16 hi/lo planes ----------
// B 32x16 bf16 operand in this code's k order (same as A): lane (half=lane>>4, l15=lane&15) holds
// K = kblk*32 + half*16 + {2j, 2j+1} at column nblk*16 + l15, packed per dword.
// dword index = ((kblk*tilesN + nblk)*32 + lane)*8 + j   -> contiguous 32B per lane.
__global__ void pack_w_k(const float* __restrict__ W, unsigned* __restrict__ Whi,
                         unsigned* __restrict__ Wlo, int K, int C, int Cpad) {
  int idx = blockIdx.x * blockDim.x + threadIdx.x;
  int tilesN = Cpad >> 4;
  int total = (K >> 5) * tilesN * 32 * 8;
  if (idx >= total) return;
  int j    = idx & 7;
  int lane = (idx >> 3) & 31;
  int blk  = idx >> 8;                 // kblk*tilesN + nblk
  int kblk = blk / tilesN, nblk = blk % tilesN;
  int half = lane >> 4, l15 = lane & 15;
  int kk  = (kblk << 5) + half * 16 + 2 * j;
  int col = (nblk << 4) + l15;
  float w0 = 0.0f, w1 = 0.0f;
  if (col < C) { w0 = W[(size_t)kk * C + col]; w1 = W[(size_t)(kk + 1) * C + col]; }
  __bf16 h0 = (__bf16)w0, h1 = (__bf16)w1;
  union { __bf16 b[2]; unsigned u; } ph, pl;
  ph.b[0] = h0;               ph.b[1] = h1;
  pl.b[0] = (__bf16)(w0 - (float)h0);
  pl.b[1] = (__bf16)(w1 - (float)h1);
  Whi[idx] = ph.u;
  Wlo[idx] = pl.u;
}

// ---------- Y[M,C] = X[M,K] @ W[K,C] via bf16x3 split WMMA (f32-accurate) ----------
// One wave per 16x16 output tile; 8 waves (256 thr) per block.
// B operand comes pre-swizzled from pack_w_k: branch-free hot loop.
__global__ void gemm_xw_wmma(const float* __restrict__ X,
                             const unsigned* __restrict__ Whi,
                             const unsigned* __restrict__ Wlo,
                             float* __restrict__ Y, int M, int K, int C, int Cpad) {
  const int lane   = threadIdx.x & 31;
  const int wid    = threadIdx.x >> 5;
  const int tilesN = Cpad >> 4;
  const int tiles  = (M >> 4) * tilesN;
  int tile = blockIdx.x * (blockDim.x >> 5) + wid;
  if (tile >= tiles) return;
  const int tm   = (tile / tilesN) << 4;
  const int nblk = tile % tilesN;
  const int tn   = nblk << 4;
  const int half = lane >> 4;      // 0: lanes 0-15, 1: lanes 16-31
  const int l15  = lane & 15;
  const int arow = tm + l15;       // A: M = lane&15
  const int bcol = tn + l15;       // B: N = lane&15
  const float* __restrict__ Xr = X + (size_t)arow * K + half * 16;

  v8f acc = {};
  for (int k0 = 0, kblk = 0; k0 < K; k0 += 32, ++kblk) {
    // ---- A 16x32 bf16 operand: one contiguous 16-float run per lane ----
    float f[16];
    *(float4*)(f + 0)  = *(const float4*)(Xr + k0 + 0);
    *(float4*)(f + 4)  = *(const float4*)(Xr + k0 + 4);
    *(float4*)(f + 8)  = *(const float4*)(Xr + k0 + 8);
    *(float4*)(f + 12) = *(const float4*)(Xr + k0 + 12);
    v16bf ah, al;
#pragma unroll
    for (int i = 0; i < 16; ++i) {
      __bf16 h = (__bf16)f[i];
      ah[i] = h;
      al[i] = (__bf16)(f[i] - (float)h);
    }
    // ---- B operand: pre-swizzled, 32B contiguous per lane ----
    const size_t base = ((size_t)(kblk * tilesN + nblk) * 32 + lane) * 8;
    v16bf bh = *(const v16bf*)(Whi + base);
    v16bf bl = *(const v16bf*)(Wlo + base);
    // bf16x3 split: (ah+al)*(bh+bl) ~= ah*bh + al*bh + ah*bl
    acc = __builtin_amdgcn_wmma_f32_16x16x32_bf16(false, al, false, bh, (short)0, acc, false, false);
    acc = __builtin_amdgcn_wmma_f32_16x16x32_bf16(false, ah, false, bl, (short)0, acc, false, false);
    acc = __builtin_amdgcn_wmma_f32_16x16x32_bf16(false, ah, false, bh, (short)0, acc, false, false);
  }
  // C/D layout: VGPR r -> row tm + r + 8*half, col tn + (lane&15)
#pragma unroll
  for (int r = 0; r < 8; ++r) {
    if (bcol < C) Y[(size_t)(tm + r + half * 8) * C + bcol] = acc[r];
  }
}

// ---------- edge pass 1: e = att . leaky_relu(xl[src] + xr[dst]); seg-max via atomic ----------
__global__ void edge_logits_k(const float* __restrict__ xl, const float* __restrict__ xr,
                              const float* __restrict__ att, const int* __restrict__ ei,
                              float* __restrict__ e, unsigned* __restrict__ menc,
                              int nE, int n, int C) {
  int eid = blockIdx.x * blockDim.x + threadIdx.x;
  if (eid >= nE + n) return;
  int src, dst;
  if (eid < nE) { src = ei[eid]; dst = ei[nE + eid]; }
  else          { src = eid - nE; dst = src; }
  const float* pl = xl + (size_t)src * C;
  const float* pr = xr + (size_t)dst * C;
  float acc = 0.0f;
  for (int c = 0; c < C; c += 4) {
    float4 a = *(const float4*)(pl + c);
    float4 b = *(const float4*)(pr + c);
    float4 t = *(const float4*)(att + c);
    float v0 = a.x + b.x; v0 = v0 > 0.0f ? v0 : 0.2f * v0;
    float v1 = a.y + b.y; v1 = v1 > 0.0f ? v1 : 0.2f * v1;
    float v2 = a.z + b.z; v2 = v2 > 0.0f ? v2 : 0.2f * v2;
    float v3 = a.w + b.w; v3 = v3 > 0.0f ? v3 : 0.2f * v3;
    acc += t.x * v0 + t.y * v1 + t.z * v2 + t.w * v3;
  }
  e[eid] = acc;
  atomicMax(&menc[dst], enc_ord(acc));
}

// ---------- edge pass 2: ex = exp(e - m[dst]); seg-sum via atomic ----------
__global__ void edge_exp_k(float* __restrict__ e, const unsigned* __restrict__ menc,
                           float* __restrict__ s, const int* __restrict__ ei, int nE, int n) {
  int eid = blockIdx.x * blockDim.x + threadIdx.x;
  if (eid >= nE + n) return;
  int dst = (eid < nE) ? ei[nE + eid] : (eid - nE);
  float ex = __expf(e[eid] - dec_ord(menc[dst]));
  e[eid] = ex;
  atomicAdd(&s[dst], ex);
}

// ---------- edge pass 3: out[dst] += (ex/s[dst]) * xl[src] ----------
__global__ void edge_scatter_k(const float* __restrict__ e, const float* __restrict__ s,
                               const float* __restrict__ xl, const int* __restrict__ ei,
                               float* __restrict__ out, int nE, int n, int C) {
  int eid = blockIdx.x * blockDim.x + threadIdx.x;
  if (eid >= nE + n) return;
  int src, dst;
  if (eid < nE) { src = ei[eid]; dst = ei[nE + eid]; }
  else          { src = eid - nE; dst = src; }
  float w = e[eid] / s[dst];
  const float* pl = xl + (size_t)src * C;
  float* po = out + (size_t)dst * C;
  for (int c = 0; c < C; c += 4) {
    float4 a = *(const float4*)(pl + c);
    atomicAdd(po + c + 0, w * a.x);
    atomicAdd(po + c + 1, w * a.y);
    atomicAdd(po + c + 2, w * a.z);
    atomicAdd(po + c + 3, w * a.w);
  }
}

// ---------- fused bias + batchnorm + relu ----------
__global__ void bn_relu_k(float* __restrict__ h, const float* __restrict__ b,
                          const float* __restrict__ g, const float* __restrict__ beta,
                          const float* __restrict__ m, const float* __restrict__ v,
                          int n, int C) {
  int i = blockIdx.x * blockDim.x + threadIdx.x;
  if (i >= n * C) return;
  int c = i % C;
  float val = h[i] + b[c];
  val = (val - m[c]) * (g[c] * rsqrtf(v[c] + 1e-5f)) + beta[c];
  h[i] = val > 0.0f ? val : 0.0f;
}

// ---------- final: out = log_softmax(h + b3) ----------
__global__ void final_lsm_k(const float* __restrict__ h, const float* __restrict__ b,
                            float* __restrict__ out, int n) {
  int i = blockIdx.x * blockDim.x + threadIdx.x;
  if (i >= n) return;
  float vals[DOUT];
  float mx = -3.4e38f;
  const float* hi = h + (size_t)i * DOUT;
#pragma unroll
  for (int c = 0; c < DOUT; ++c) { float v = hi[c] + b[c]; vals[c] = v; mx = fmaxf(mx, v); }
  float sum = 0.0f;
#pragma unroll
  for (int c = 0; c < DOUT; ++c) sum += __expf(vals[c] - mx);
  float ls = __logf(sum) + mx;
  float* oi = out + (size_t)i * DOUT;
#pragma unroll
  for (int c = 0; c < DOUT; ++c) oi[c] = vals[c] - ls;
}

extern "C" void kernel_launch(void* const* d_in, const int* in_sizes, int n_in,
                              void* d_out, int out_size, void* d_ws, size_t ws_size,
                              hipStream_t stream) {
  (void)in_sizes; (void)n_in; (void)out_size; (void)ws_size;
  const float* x    = (const float*)d_in[0];
  const int*   ei   = (const int*)d_in[1];
  const float* Wl1  = (const float*)d_in[2];
  const float* Wr1  = (const float*)d_in[3];
  const float* att1 = (const float*)d_in[4];
  const float* b1   = (const float*)d_in[5];
  const float* g1   = (const float*)d_in[6];
  const float* be1  = (const float*)d_in[7];
  const float* m1   = (const float*)d_in[8];
  const float* v1   = (const float*)d_in[9];
  const float* Wl2  = (const float*)d_in[10];
  const float* Wr2  = (const float*)d_in[11];
  const float* att2 = (const float*)d_in[12];
  const float* b2   = (const float*)d_in[13];
  const float* g2   = (const float*)d_in[14];
  const float* be2  = (const float*)d_in[15];
  const float* m2   = (const float*)d_in[16];
  const float* v2   = (const float*)d_in[17];
  const float* Wl3  = (const float*)d_in[18];
  const float* Wr3  = (const float*)d_in[19];
  const float* att3 = (const float*)d_in[20];
  const float* b3   = (const float*)d_in[21];
  float* out = (float*)d_out;

  // workspace layout (floats)
  float* ws   = (float*)d_ws;
  float* xl   = ws;                                    // [NN, 64] (layer3 uses [NN,40])
  float* xr   = xl + (size_t)NN * DH;                  // [NN, 64]
  float* ebuf = xr + (size_t)NN * DH;                  // [NE + NN]
  unsigned* menc = (unsigned*)(ebuf + (size_t)(NE + NN));  // [NN]
  float* s    = (float*)(menc + NN);                   // [NN]
  float* h1   = s + NN;                                // [NN, 64] act1 / layer3 out
  float* h2   = h1 + (size_t)NN * DH;                  // [NN, 64] act2
  // packed W planes (max: layer1 = (128/32)*(64/16)*256 = 4096 dwords each)
  unsigned* Whi = (unsigned*)(h2 + (size_t)NN * DH);   // [4096]
  unsigned* Wlo = Whi + 4096;                          // [4096]

  const int TB = 256;
  dim3 blk(TB);
  const int gridE  = (NE + NN + TB - 1) / TB;
  const int gridNC = (NN * DH + TB - 1) / TB;
  const int gridNO = (NN * DOUT + TB - 1) / TB;
  const int gridN  = (NN + TB - 1) / TB;
  auto gemmGrid = [](int M, int Cpad) { int t = (M / 16) * (Cpad / 16); return (t + 7) / 8; };
  auto packGrid = [](int K, int Cpad) { int t = (K / 32) * (Cpad / 16) * 256; return (t + 255) / 256; };

  // ---------------- Layer 1: x[N,128] -> h1[N,64] ----------------
  init_layer_k<<<gridNC, blk, 0, stream>>>(menc, s, h1, NN, DH);
  pack_w_k<<<packGrid(DIN, 64), blk, 0, stream>>>(Wl1, Whi, Wlo, DIN, DH, 64);
  gemm_xw_wmma<<<gemmGrid(NN, 64), blk, 0, stream>>>(x, Whi, Wlo, xl, NN, DIN, DH, 64);
  pack_w_k<<<packGrid(DIN, 64), blk, 0, stream>>>(Wr1, Whi, Wlo, DIN, DH, 64);
  gemm_xw_wmma<<<gemmGrid(NN, 64), blk, 0, stream>>>(x, Whi, Wlo, xr, NN, DIN, DH, 64);
  edge_logits_k<<<gridE, blk, 0, stream>>>(xl, xr, att1, ei, ebuf, menc, NE, NN, DH);
  edge_exp_k<<<gridE, blk, 0, stream>>>(ebuf, menc, s, ei, NE, NN);
  edge_scatter_k<<<gridE, blk, 0, stream>>>(ebuf, s, xl, ei, h1, NE, NN, DH);
  bn_relu_k<<<gridNC, blk, 0, stream>>>(h1, b1, g1, be1, m1, v1, NN, DH);

  // ---------------- Layer 2: h1[N,64] -> h2[N,64] ----------------
  init_layer_k<<<gridNC, blk, 0, stream>>>(menc, s, h2, NN, DH);
  pack_w_k<<<packGrid(DH, 64), blk, 0, stream>>>(Wl2, Whi, Wlo, DH, DH, 64);
  gemm_xw_wmma<<<gemmGrid(NN, 64), blk, 0, stream>>>(h1, Whi, Wlo, xl, NN, DH, DH, 64);
  pack_w_k<<<packGrid(DH, 64), blk, 0, stream>>>(Wr2, Whi, Wlo, DH, DH, 64);
  gemm_xw_wmma<<<gemmGrid(NN, 64), blk, 0, stream>>>(h1, Whi, Wlo, xr, NN, DH, DH, 64);
  edge_logits_k<<<gridE, blk, 0, stream>>>(xl, xr, att2, ei, ebuf, menc, NE, NN, DH);
  edge_exp_k<<<gridE, blk, 0, stream>>>(ebuf, menc, s, ei, NE, NN);
  edge_scatter_k<<<gridE, blk, 0, stream>>>(ebuf, s, xl, ei, h2, NE, NN, DH);
  bn_relu_k<<<gridNC, blk, 0, stream>>>(h2, b2, g2, be2, m2, v2, NN, DH);

  // ---------------- Layer 3: h2[N,64] -> h1[N,40] -> log_softmax ----------------
  init_layer_k<<<gridNO, blk, 0, stream>>>(menc, s, h1, NN, DOUT);
  pack_w_k<<<packGrid(DH, 48), blk, 0, stream>>>(Wl3, Whi, Wlo, DH, DOUT, 48);
  gemm_xw_wmma<<<gemmGrid(NN, 48), blk, 0, stream>>>(h2, Whi, Wlo, xl, NN, DH, DOUT, 48);
  pack_w_k<<<packGrid(DH, 48), blk, 0, stream>>>(Wr3, Whi, Wlo, DH, DOUT, 48);
  gemm_xw_wmma<<<gemmGrid(NN, 48), blk, 0, stream>>>(h2, Whi, Wlo, xr, NN, DH, DOUT, 48);
  edge_logits_k<<<gridE, blk, 0, stream>>>(xl, xr, att3, ei, ebuf, menc, NE, NN, DOUT);
  edge_exp_k<<<gridE, blk, 0, stream>>>(ebuf, menc, s, ei, NE, NN);
  edge_scatter_k<<<gridE, blk, 0, stream>>>(ebuf, s, xl, ei, h1, NE, NN, DOUT);
  final_lsm_k<<<gridN, blk, 0, stream>>>(h1, b3, out, NN);
}
